// DGLJTMPN_13993003450825
// MI455X (gfx1250) — matrix operand order repaired and checked by reading, not compiled, not run on hardware
//
#include <hip/hip_runtime.h>
#include <hip/hip_bf16.h>

typedef __bf16 bf16_t;
typedef bf16_t v16bf __attribute__((ext_vector_type(16)));
typedef bf16_t v4bf  __attribute__((ext_vector_type(4)));
typedef float  v8f   __attribute__((ext_vector_type(8)));
typedef float  v4f   __attribute__((ext_vector_type(4)));

#define HID 384
#define NT  24           // 384/16 column tiles
#define AFD 35
#define BFD 5

// ---------------------------------------------------------------------------
// Weight packing: fragment-ordered bf16 layout for V_WMMA_F32_16X16X32_BF16 B
// block b = kt*NT + nt holds a 32x16 (KxN) tile as 512 bf16:
//   lane l (0..31), elem m (0..15):  K = kt*32 + (l<16?0:16) + m,  N = nt*16 + (l&15)
// ---------------------------------------------------------------------------
__global__ void pack_w_kernel(const float* __restrict__ W, bf16_t* __restrict__ out,
                              int K, int KPAD) {
    int idx = blockIdx.x * blockDim.x + threadIdx.x;
    int total = KPAD * HID;
    if (idx >= total) return;
    int b = idx >> 9;            // /512
    int r = idx & 511;
    int l = r >> 4;              // lane
    int m = r & 15;              // element
    int kt = b / NT, nt = b - kt * NT;
    int n = nt * 16 + (l & 15);
    int k = kt * 32 + ((l < 16) ? 0 : 16) + m;
    float v = (k < K) ? W[(size_t)k * HID + n] : 0.f;
    out[idx] = (bf16_t)v;
}

__global__ void zero_f_kernel(float* __restrict__ p, long n) {
    long i = blockIdx.x * (long)blockDim.x + threadIdx.x;
    if (i < n) p[i] = 0.f;
}

__global__ void copy_f4_kernel(float4* __restrict__ dst, const float4* __restrict__ src, long n4) {
    long i = blockIdx.x * (long)blockDim.x + threadIdx.x;
    if (i < n4) dst[i] = src[i];
}

// atomic scatter-add of [rows, cols] fp32 into dst rows idx[r]; 4 columns/thread.
// Target buffer is L2-resident on MI455X (192MB L2 >> 154MB node state), so the
// global_atomic_add_f32 traffic resolves in L2 without touching HBM.
__global__ void scatter_rows4_kernel(const float* __restrict__ src, const int* __restrict__ idx,
                                     float* __restrict__ dst, long total4, int cols) {
    long i = blockIdx.x * (long)blockDim.x + threadIdx.x;
    if (i >= total4) return;
    const int c4 = cols >> 2;
    long r = i / c4;
    int  c = (int)(i - r * c4) << 2;
    const v4f v = *(const v4f*)(src + r * cols + c);
    float* d = dst + (size_t)idx[r] * cols + c;
    atomicAdd(d + 0, v[0]);
    atomicAdd(d + 1, v[1]);
    atomicAdd(d + 2, v[2]);
    atomicAdd(d + 3, v[3]);
}

__global__ void count_kernel(const int* __restrict__ gid, float* __restrict__ cnt, int n) {
    int i = blockIdx.x * blockDim.x + threadIdx.x;
    if (i < n) atomicAdd(&cnt[gid[i]], 1.f);
}

__global__ void finalize_kernel(const float* __restrict__ sums, const float* __restrict__ cnt,
                                float* __restrict__ out, long total) {
    long i = blockIdx.x * (long)blockDim.x + threadIdx.x;
    if (i >= total) return;
    long g = i / HID;
    out[i] = sums[i] / fmaxf(cnt[g], 1.f);
}

// ---------------------------------------------------------------------------
// Fused GEMM: rows x KPAD (bf16, staged in LDS) @ KPAD x 384 (packed bf16)
// MODE 0: edge-input  A = [x[src[e]] ; bond_x[e] ; pad]      ; out msg_input + relu msg
// MODE 1: msg step    A = node_in[src[e]] - m[e^1]           ; out m_new = relu(msg_input + AB)
// MODE 2: readout     A = [x[n] ; m_node[n] ; pad]           ; h = relu(AB + b_o) -> atomic graph sum
// 512 threads = 16 waves; wave w: rows (w>>2)*16..+15, cols ((w&3)*6)*16 .. +95
// LDS row stride = KPAD+8 bf16 (196 dwords mod 64 = 4) -> conflict-free ds_load_b128.
// ---------------------------------------------------------------------------
template <int MODE, int KPAD>
__global__ __launch_bounds__(512)
void gemm_kernel(int nrows,
                 const float* __restrict__ x,
                 const float* __restrict__ bond,
                 const int*   __restrict__ esrc,
                 const float* __restrict__ nodeIn,
                 const float* __restrict__ mcur,
                 const float* __restrict__ msgIn,
                 const bf16_t* __restrict__ Wp,
                 const float* __restrict__ bo,
                 const int*   __restrict__ gid,
                 float* __restrict__ out0,
                 float* __restrict__ out1) {
    constexpr int LSTR = KPAD + 8;       // padded LDS row stride (bf16 elements)
    __shared__ bf16_t tl[64 * LSTR];
    const int tid  = threadIdx.x;
    const int base = blockIdx.x * 64;

    // Phase 1: build bf16 A-tile in LDS
    if (MODE == 1) {
        // hot path: float4 loads, packed 4xbf16 (b64) LDS stores
        constexpr int KP4 = KPAD / 4;
        for (int idx = tid; idx < 64 * KP4; idx += 512) {
            int r  = idx / KP4;
            int k  = (idx - r * KP4) << 2;
            int row = base + r;
            v4f v = {0.f, 0.f, 0.f, 0.f};
            if (row < nrows) {
                const v4f a = *(const v4f*)(nodeIn + (size_t)esrc[row] * HID + k);
                const v4f b = *(const v4f*)(mcur + (size_t)(row ^ 1) * HID + k);
                v = a - b;
            }
            *(v4bf*)(&tl[(size_t)r * LSTR + k]) = __builtin_convertvector(v, v4bf);
        }
        // pull the epilogue stream into cache while the WMMA loop runs
        const char* p = (const char*)(msgIn + (size_t)base * HID);
        __builtin_prefetch(p + tid * 192, 0, 1);
    } else {
        for (int idx = tid; idx < 64 * KPAD; idx += 512) {
            int r = idx / KPAD;
            int k = idx - r * KPAD;
            int row = base + r;
            float v = 0.f;
            if (row < nrows) {
                if (MODE == 0) {
                    if (k < AFD)            v = x[(size_t)esrc[row] * AFD + k];
                    else if (k < AFD + BFD) v = bond[(size_t)row * BFD + (k - AFD)];
                } else {
                    if (k < AFD)            v = x[(size_t)row * AFD + k];
                    else if (k < AFD + HID) v = nodeIn[(size_t)row * HID + (k - AFD)];
                }
            }
            tl[(size_t)r * LSTR + k] = (bf16_t)v;
        }
    }
    __syncthreads();

    const int lane  = tid & 31;
    const int w     = tid >> 5;
    const int rr    = (w >> 2) * 16;     // LDS row-tile base
    const int cq    = (w & 3) * 6;       // first N-tile
    const int kbase = (lane < 16) ? 0 : 16;
    const int arow  = rr + (lane & 15);

    v8f acc[6] = {};
    for (int kt = 0; kt < KPAD / 32; ++kt) {
        // A fragment (contiguous k order matching packed B: lanes 0-15 K 0-15, lanes 16-31 K 16-31)
        v16bf a;
        const bf16_t* ap = &tl[(size_t)arow * LSTR + kt * 32 + kbase];
        #pragma unroll
        for (int i = 0; i < 8; ++i)  a[i] = ap[i];
        #pragma unroll
        for (int i = 8; i < 16; ++i) a[i] = ap[i];
        #pragma unroll
        for (int t = 0; t < 6; ++t) {
            const v16bf b = *(const v16bf*)(Wp + ((size_t)(kt * NT + cq + t) << 9) + lane * 16);
            acc[t] = __builtin_amdgcn_wmma_f32_16x16x32_bf16(
                false, a, false, b, (short)0, acc[t], false, false);
        }
    }

    // Epilogue (C/D layout: col = lane&15, row = vgpr + (lane<16?0:8))
    const int roff = (lane < 16) ? 0 : 8;
    #pragma unroll
    for (int t = 0; t < 6; ++t) {
        const int n = (cq + t) * 16 + (lane & 15);
        #pragma unroll
        for (int r = 0; r < 8; ++r) {
            const int row = base + rr + r + roff;
            if (row >= nrows) continue;
            const float v = acc[t][r];
            if (MODE == 0) {
                out0[(size_t)row * HID + n] = v;
                out1[(size_t)row * HID + n] = fmaxf(v, 0.f);
            } else if (MODE == 1) {
                out0[(size_t)row * HID + n] = fmaxf(msgIn[(size_t)row * HID + n] + v, 0.f);
            } else {
                const float h = fmaxf(v + bo[n], 0.f);
                atomicAdd(&out0[(size_t)gid[row] * HID + n], h);
            }
        }
    }
}

// ---------------------------------------------------------------------------
extern "C" void kernel_launch(void* const* d_in, const int* in_sizes, int n_in,
                              void* d_out, int out_size, void* d_ws, size_t ws_size,
                              hipStream_t stream) {
    const float* x          = (const float*)d_in[0];
    const float* bond_x     = (const float*)d_in[1];
    const int*   edge_src   = (const int*)d_in[2];
    const int*   edge_dst   = (const int*)d_in[3];
    const float* tree_alpha = (const float*)d_in[4];
    const int*   tree_tgt   = (const int*)d_in[5];
    const int*   graph_ids  = (const int*)d_in[6];
    const float* W_i        = (const float*)d_in[8];
    const float* W_h        = (const float*)d_in[9];
    const float* W_o        = (const float*)d_in[10];
    const float* b_o        = (const float*)d_in[11];

    const int N = in_sizes[0] / AFD;
    const int E = in_sizes[2];
    const int T = in_sizes[4] / HID;
    const int G = out_size / HID;

    // workspace carve-out (256B aligned)
    auto up = [](size_t v) { return (v + 255) & ~(size_t)255; };
    char* p = (char*)d_ws;
    auto take = [&](size_t bytes) { char* q = p; p += up(bytes); return q; };
    float*  nodeAlpha = (float*)take((size_t)N * HID * 4);
    float*  nodeIn    = (float*)take((size_t)N * HID * 4);
    float*  msgInput  = (float*)take((size_t)E * HID * 4);
    float*  mA        = (float*)take((size_t)E * HID * 4);
    float*  mB        = (float*)take((size_t)E * HID * 4);
    bf16_t* pWi       = (bf16_t*)take((size_t)64  * HID * 2);
    bf16_t* pWh       = (bf16_t*)take((size_t)384 * HID * 2);
    bf16_t* pWo       = (bf16_t*)take((size_t)448 * HID * 2);
    float*  sums      = (float*)take((size_t)G * HID * 4);
    float*  cnts      = (float*)take((size_t)G * 4);
    (void)ws_size; (void)n_in;

    auto g1 = [](long n, int t) { return (int)((n + t - 1) / t); };

    // node_alpha = scatter_add(tree_alpha)
    zero_f_kernel<<<g1((long)N * HID, 256), 256, 0, stream>>>(nodeAlpha, (long)N * HID);
    scatter_rows4_kernel<<<g1((long)T * HID / 4, 256), 256, 0, stream>>>(
        tree_alpha, tree_tgt, nodeAlpha, (long)T * HID / 4, HID);

    // pack weights to bf16 fragment layout
    pack_w_kernel<<<g1(64  * HID, 256), 256, 0, stream>>>(W_i, pWi, AFD + BFD, 64);
    pack_w_kernel<<<g1(384 * HID, 256), 256, 0, stream>>>(W_h, pWh, HID, 384);
    pack_w_kernel<<<g1(448 * HID, 256), 256, 0, stream>>>(W_o, pWo, AFD + HID, 448);

    // msg_input = [x[src];bond] @ W_i ; m0 = relu(msg_input)
    gemm_kernel<0, 64><<<g1(E, 64), 512, 0, stream>>>(
        E, x, bond_x, edge_src, nullptr, nullptr, nullptr, pWi, nullptr, nullptr,
        msgInput, mA);

    float* cur = mA;
    float* nxt = mB;
    for (int it = 0; it < 3; ++it) {   // DEPTH-1
        // node_in = node_alpha + segment_sum(m over edge_dst)   (L2-resident atomics)
        copy_f4_kernel<<<g1((long)N * HID / 4, 256), 256, 0, stream>>>(
            (float4*)nodeIn, (const float4*)nodeAlpha, (long)N * HID / 4);
        scatter_rows4_kernel<<<g1((long)E * HID / 4, 256), 256, 0, stream>>>(
            cur, edge_dst, nodeIn, (long)E * HID / 4, HID);
        // m_new = relu(msg_input + (node_in[src] - m[rev]) @ W_h)
        gemm_kernel<1, 384><<<g1(E, 64), 512, 0, stream>>>(
            E, nullptr, nullptr, edge_src, nodeIn, cur, msgInput, pWh, nullptr, nullptr,
            nxt, nullptr);
        float* t2 = cur; cur = nxt; nxt = t2;
    }

    // m_node = node_alpha + segment_sum(msg over edge_dst)
    copy_f4_kernel<<<g1((long)N * HID / 4, 256), 256, 0, stream>>>(
        (float4*)nodeIn, (const float4*)nodeAlpha, (long)N * HID / 4);
    scatter_rows4_kernel<<<g1((long)E * HID / 4, 256), 256, 0, stream>>>(
        cur, edge_dst, nodeIn, (long)E * HID / 4, HID);

    // h = relu([x;m_node]@W_o + b_o)  -> per-graph sums (fused atomic reduction)
    zero_f_kernel<<<g1((long)G * HID, 256), 256, 0, stream>>>(sums, (long)G * HID);
    zero_f_kernel<<<g1(G, 256), 256, 0, stream>>>(cnts, G);
    gemm_kernel<2, 448><<<g1(N, 64), 512, 0, stream>>>(
        N, x, nullptr, nullptr, nodeIn, nullptr, nullptr, pWo, b_o, graph_ids,
        sums, nullptr);
    count_kernel<<<g1(N, 256), 256, 0, stream>>>(graph_ids, cnts, N);

    finalize_kernel<<<g1((long)G * HID, 256), 256, 0, stream>>>(
        sums, cnts, (float*)d_out, (long)G * HID);
}
